// Qwen3_5GatedDeltaNet_79010218377270
// MI455X (gfx1250) — matrix operand order repaired and checked by reading, not compiled, not run on hardware
//
#include <hip/hip_runtime.h>
#include <hip/hip_bf16.h>
#include <cstdint>
#include <cstddef>

// ---------- types for CDNA5 WMMA ----------
typedef __attribute__((ext_vector_type(16))) __bf16 v16bf;
typedef __attribute__((ext_vector_type(8)))  __bf16 v8bf;
typedef __attribute__((ext_vector_type(4)))  __bf16 v4bf;
typedef __attribute__((ext_vector_type(8)))  float  v8f;
typedef __attribute__((ext_vector_type(4)))  float  v4f;

// Problem constants (B=1)
#define S_LEN   1024
#define H_DIM   2048
#define NKH     16
#define NVH     32
#define DKD     128
#define DVD     128
#define KEYD    2048      // NK*DK
#define VALD    4096      // NV*DV
#define CONVD   8192      // 2*KEYD + VALD
#define ZBAD    4160      // VALD + 2*NV
#define EPSF    1e-6f

// GEMM tiling
#define BM 64
#define BN 256
#define BK 32

// ---------- global-address-space helpers ----------
// Force GLOBAL_LOAD/GLOBAL_STORE instead of FLAT (FLAT ties up DScnt and the
// LDS path per CDNA5 ISA §10.1; Global uses only LOADcnt/STOREcnt).
// Use trivial ext_vector types so AS(1) deref is a plain vector load/store.
__device__ __forceinline__ v4f ldg4(const float* p) {
    return *(const __attribute__((address_space(1))) v4f*)p;
}
__device__ __forceinline__ float ldg(const float* p) {
    return *(const __attribute__((address_space(1))) float*)p;
}
__device__ __forceinline__ void stg(float* p, float v) {
    *(__attribute__((address_space(1))) float*)p = v;
}
__device__ __forceinline__ void stg4(float* p, v4f v) {
    *(__attribute__((address_space(1))) v4f*)p = v;
}

__device__ __forceinline__ void cvt_store4(__bf16* p, v4f f) {
    v4bf o;
    o[0] = (__bf16)f[0]; o[1] = (__bf16)f[1];
    o[2] = (__bf16)f[2]; o[3] = (__bf16)f[3];
    *(v4bf*)p = o;                       // 2x v_cvt_pk_bf16_f32 + ds_store_b64
}

// =====================================================================
// GEMM: C[M,N] = A[M,K] (f32, row-major) x W[N,K]^T (f32, row-major)
// bf16 WMMA 16x16x32, f32 accumulate. Block tile 64x256, 8 waves,
// each wave 32x64 (2x4 WMMA tiles). Register-pipelined global->LDS
// staging with f32->bf16 conversion.
// =====================================================================
__global__ __launch_bounds__(256) void gemm_bf16_wmma(
    const float* __restrict__ A, const float* __restrict__ W,
    float* __restrict__ C, int M, int N, int K)
{
    __shared__ __bf16 sA[BM * BK];    // [row][k]   4 KB
    __shared__ __bf16 sB[BN * BK];    // [col][k]  16 KB

    const int tid  = threadIdx.x;
    const int lane = tid & 31;
    const int wave = tid >> 5;
    const int bm   = blockIdx.y * BM;
    const int bn   = blockIdx.x * BN;
    const int wm   = (wave & 1) * 32;     // wave row offset in block tile
    const int wn   = (wave >> 1) * 64;    // wave col offset in block tile
    const int half = lane >> 4;           // 0: lanes 0-15, 1: lanes 16-31
    const int l16  = lane & 15;

    v8f acc[2][4];
    #pragma unroll
    for (int i = 0; i < 2; ++i)
        #pragma unroll
        for (int j = 0; j < 4; ++j)
            #pragma unroll
            for (int e = 0; e < 8; ++e) acc[i][j][e] = 0.0f;

    const v4f f4z = {0.f, 0.f, 0.f, 0.f};
    v4f ra[2], rb[8];

    // ---- prologue: stage k0 = 0 tile into registers, then LDS ----
    #pragma unroll
    for (int i = 0; i < 2; ++i) {
        int q = tid + i * 256;                 // 512 quads of A
        ra[i] = ldg4(&A[(size_t)(bm + (q >> 3)) * K + (q & 7) * 4]);
    }
    #pragma unroll
    for (int i = 0; i < 8; ++i) {
        int q = tid + i * 256;                 // 2048 quads of B
        int gn = bn + (q >> 3);
        rb[i] = (gn < N) ? ldg4(&W[(size_t)gn * K + (q & 7) * 4]) : f4z;
    }
    #pragma unroll
    for (int i = 0; i < 2; ++i) {
        int q = tid + i * 256;
        cvt_store4(&sA[(q >> 3) * BK + (q & 7) * 4], ra[i]);
    }
    #pragma unroll
    for (int i = 0; i < 8; ++i) {
        int q = tid + i * 256;
        cvt_store4(&sB[(q >> 3) * BK + (q & 7) * 4], rb[i]);
    }
    __syncthreads();

    for (int k0 = 0;;) {
        const int knext = k0 + BK;
        // ---- issue next tile's global loads (overlap with WMMA below) ----
        if (knext < K) {
            if (knext + BK < K)   // speculative prefetch 2 tiles ahead
                __builtin_prefetch(&A[(size_t)(bm + (tid & 63)) * K + knext + BK], 0, 1);
            #pragma unroll
            for (int i = 0; i < 2; ++i) {
                int q = tid + i * 256;
                ra[i] = ldg4(&A[(size_t)(bm + (q >> 3)) * K + knext + (q & 7) * 4]);
            }
            #pragma unroll
            for (int i = 0; i < 8; ++i) {
                int q = tid + i * 256;
                int gn = bn + (q >> 3);
                rb[i] = (gn < N) ? ldg4(&W[(size_t)gn * K + knext + (q & 7) * 4]) : f4z;
            }
        }

        // ---- fragments per CDNA5 16-bit WMMA VGPR layouts ----
        // A 16x32: lane<16 -> K {0..7,16..23}; lane>=16 -> K {8..15,24..31}
        // B 32x16: lane<16 -> K {0..7,16..23}; lane>=16 -> K {8..15,24..31}
        v16bf afr[2], bfr[4];
        #pragma unroll
        for (int t = 0; t < 2; ++t) {
            int row  = wm + t * 16 + l16;
            int base = half * 8;
            v8bf lo = *(const v8bf*)&sA[row * BK + base];
            v8bf hi = *(const v8bf*)&sA[row * BK + base + 16];
            #pragma unroll
            for (int e = 0; e < 8; ++e) { afr[t][e] = lo[e]; afr[t][e + 8] = hi[e]; }
        }
        #pragma unroll
        for (int t = 0; t < 4; ++t) {
            int col = wn + t * 16 + l16;
            int kb  = half * 8;
            v8bf lo = *(const v8bf*)&sB[col * BK + kb];
            v8bf hi = *(const v8bf*)&sB[col * BK + kb + 16];
            #pragma unroll
            for (int e = 0; e < 8; ++e) { bfr[t][e] = lo[e]; bfr[t][e + 8] = hi[e]; }
        }

        #pragma unroll
        for (int tm = 0; tm < 2; ++tm)
            #pragma unroll
            for (int tn = 0; tn < 4; ++tn)
                acc[tm][tn] = __builtin_amdgcn_wmma_f32_16x16x32_bf16(
                    false, afr[tm], false, bfr[tn], (short)0, acc[tm][tn],
                    false, false);

        __syncthreads();                 // all waves done reading LDS
        if (knext >= K) break;

        // ---- commit next tile to LDS ----
        #pragma unroll
        for (int i = 0; i < 2; ++i) {
            int q = tid + i * 256;
            cvt_store4(&sA[(q >> 3) * BK + (q & 7) * 4], ra[i]);
        }
        #pragma unroll
        for (int i = 0; i < 8; ++i) {
            int q = tid + i * 256;
            cvt_store4(&sB[(q >> 3) * BK + (q & 7) * 4], rb[i]);
        }
        __syncthreads();
        k0 = knext;
    }

    // C/D layout: VGPR j -> (M = j + 8*half, N = l16) within each 16x16 tile
    #pragma unroll
    for (int tm = 0; tm < 2; ++tm)
        #pragma unroll
        for (int tn = 0; tn < 4; ++tn) {
            int rbase = bm + wm + tm * 16 + half * 8;
            int col   = bn + wn + tn * 16 + l16;
            if (col < N) {
                #pragma unroll
                for (int j = 0; j < 8; ++j)
                    stg(&C[(size_t)(rbase + j) * N + col], acc[tm][tn][j]);
            }
        }
}

// =====================================================================
// Depthwise causal conv (K=4) + SiLU, 4 channels per thread (vector I/O)
// =====================================================================
__global__ void conv_silu_kernel(const float* __restrict__ qkv,
                                 const float* __restrict__ conv_w,
                                 float* __restrict__ out)
{
    int idx = (blockIdx.x * blockDim.x + threadIdx.x) * 4;   // s*8192 + c
    int c = idx & (CONVD - 1);
    int s = idx >> 13;
    v4f w[4];
    #pragma unroll
    for (int j = 0; j < 4; ++j) w[j] = ldg4(&conv_w[(c + j) * 4]);
    float a[4] = {0.f, 0.f, 0.f, 0.f};
    #pragma unroll
    for (int i = 0; i < 4; ++i) {
        int sp = s - 3 + i;
        if (sp >= 0) {
            v4f x = ldg4(&qkv[(size_t)sp * CONVD + c]);
            a[0] = fmaf(x[0], w[0][i], a[0]);
            a[1] = fmaf(x[1], w[1][i], a[1]);
            a[2] = fmaf(x[2], w[2][i], a[2]);
            a[3] = fmaf(x[3], w[3][i], a[3]);
        }
    }
    v4f r;
    r[0] = a[0] * (1.0f / (1.0f + __expf(-a[0])));
    r[1] = a[1] * (1.0f / (1.0f + __expf(-a[1])));
    r[2] = a[2] * (1.0f / (1.0f + __expf(-a[2])));
    r[3] = a[3] * (1.0f / (1.0f + __expf(-a[3])));
    stg4(&out[idx], r);
}

// =====================================================================
// Per-head L2 norm of q (with DK^-0.5) and k, 128 threads = one head vec
// =====================================================================
__global__ __launch_bounds__(128) void qk_norm_kernel(
    const float* __restrict__ act, float* __restrict__ qn, float* __restrict__ kn)
{
    __shared__ float red[128];
    int s = blockIdx.x;
    int hh = blockIdx.y;
    int t = threadIdx.x;
    bool isK = hh >= NKH;
    int h = isK ? hh - NKH : hh;
    size_t base = (size_t)s * CONVD + (isK ? KEYD : 0) + (size_t)h * DKD;
    float x = ldg(&act[base + t]);
    red[t] = x * x;
    __syncthreads();
    for (int off = 64; off > 0; off >>= 1) {
        if (t < off) red[t] += red[t + off];
        __syncthreads();
    }
    float y = x * rsqrtf(red[0] + EPSF);
    size_t o = ((size_t)s * NKH + h) * DKD + t;
    if (isK) stg(&kn[o], y);
    else     stg(&qn[o], y * 0.08838834764831845f);   // 128^-0.5
}

// =====================================================================
// Gate precompute: eg = exp(-exp(A_log)*softplus(a+dt_bias)), beta = sigmoid(b)
// =====================================================================
__global__ void gate_kernel(const float* __restrict__ zba,
                            const float* __restrict__ A_log,
                            const float* __restrict__ dt_bias,
                            float* __restrict__ eg, float* __restrict__ bt)
{
    int idx = blockIdx.x * blockDim.x + threadIdx.x;   // s*32 + n
    int n = idx & 31;
    int s = idx >> 5;
    const float* row = zba + (size_t)s * ZBAD;
    float b = ldg(&row[VALD + n]);
    float a = ldg(&row[VALD + NVH + n]);
    float x = a + ldg(&dt_bias[n]);
    float sp = (x > 20.0f) ? x : log1pf(__expf(x));
    float g = -__expf(ldg(&A_log[n])) * sp;
    stg(&eg[idx], __expf(g));
    stg(&bt[idx], 1.0f / (1.0f + __expf(-b)));
}

// =====================================================================
// Sequential delta-rule scan, parallelized: grid (32 heads x 4 v-chunks),
// 128 threads = 32 v-columns x 4 k-quarters. Each thread holds 32 state
// elements in registers; k-reductions via wave32 shfl_xor (lanes 1,2);
// dual accumulators shorten the serial FMA chain to 16.
// =====================================================================
__global__ __launch_bounds__(128) void scan_kernel(
    const float* __restrict__ qn, const float* __restrict__ kn,
    const float* __restrict__ act, const float* __restrict__ eg,
    const float* __restrict__ bt, float* __restrict__ core)
{
    __shared__ float qs[128], ks[128], vs[32];
    __shared__ float sg, sb;
    const int n   = blockIdx.x;          // value head
    const int vc  = blockIdx.y;          // v-chunk (32 columns)
    const int t   = threadIdx.x;
    const int vl  = t >> 2;              // local v column 0..31
    const int kq  = t & 3;               // k quarter
    const int kb  = kq * 32;
    const int hq  = n >> 1;              // GQA head mapping

    float st[32];
    #pragma unroll
    for (int i = 0; i < 32; ++i) st[i] = 0.0f;

    for (int s = 0; s < S_LEN; ++s) {
        size_t qbs = ((size_t)s * NKH + hq) * DKD;
        qs[t] = ldg(&qn[qbs + t]);
        ks[t] = ldg(&kn[qbs + t]);
        if (t < 32)
            vs[t] = ldg(&act[(size_t)s * CONVD + 2 * KEYD + (size_t)n * DVD + vc * 32 + t]);
        if (t == 0) { sg = ldg(&eg[s * NVH + n]); sb = ldg(&bt[s * NVH + n]); }
        __syncthreads();

        float e = sg, btv = sb;
        float kv0 = 0.f, kv1 = 0.f;
        #pragma unroll
        for (int i = 0; i < 32; i += 2) {
            st[i]     *= e; kv0 = fmaf(st[i],     ks[kb + i],     kv0);
            st[i + 1] *= e; kv1 = fmaf(st[i + 1], ks[kb + i + 1], kv1);
        }
        float kv = kv0 + kv1;
        kv += __shfl_xor(kv, 1, 32);
        kv += __shfl_xor(kv, 2, 32);

        float delta = (vs[vl] - kv) * btv;

        float o0 = 0.f, o1 = 0.f;
        #pragma unroll
        for (int i = 0; i < 32; i += 2) {
            st[i]     = fmaf(ks[kb + i],     delta, st[i]);
            o0        = fmaf(st[i],          qs[kb + i],     o0);
            st[i + 1] = fmaf(ks[kb + i + 1], delta, st[i + 1]);
            o1        = fmaf(st[i + 1],      qs[kb + i + 1], o1);
        }
        float o = o0 + o1;
        o += __shfl_xor(o, 1, 32);
        o += __shfl_xor(o, 2, 32);

        if (kq == 0)
            stg(&core[(size_t)s * VALD + (size_t)n * DVD + vc * 32 + vl], o);
        __syncthreads();
    }
}

// =====================================================================
// gated = core * silu(z); RMSNorm over DV=128; * norm_w
// =====================================================================
__global__ __launch_bounds__(128) void gated_norm_kernel(
    const float* __restrict__ core, const float* __restrict__ zba,
    const float* __restrict__ norm_w, float* __restrict__ gn)
{
    __shared__ float red[128];
    int s = blockIdx.x, n = blockIdx.y, t = threadIdx.x;
    float z = ldg(&zba[(size_t)s * ZBAD + (size_t)n * DVD + t]);
    float c = ldg(&core[(size_t)s * VALD + (size_t)n * DVD + t]);
    float gated = c * (z * (1.0f / (1.0f + __expf(-z))));
    red[t] = gated * gated;
    __syncthreads();
    for (int off = 64; off > 0; off >>= 1) {
        if (t < off) red[t] += red[t + off];
        __syncthreads();
    }
    float var = red[0] * (1.0f / 128.0f);
    stg(&gn[(size_t)s * VALD + (size_t)n * DVD + t],
        gated * rsqrtf(var + EPSF) * ldg(&norm_w[t]));
}

// =====================================================================
extern "C" void kernel_launch(void* const* d_in, const int* in_sizes, int n_in,
                              void* d_out, int out_size, void* d_ws, size_t ws_size,
                              hipStream_t stream)
{
    const float* hidden  = (const float*)d_in[0];   // (1,1024,2048)
    const float* W_qkv   = (const float*)d_in[1];   // (8192,2048)
    const float* W_zba   = (const float*)d_in[2];   // (4160,2048)
    const float* conv_w  = (const float*)d_in[3];   // (8192,4)
    const float* A_log   = (const float*)d_in[4];   // (32,)
    const float* dt_bias = (const float*)d_in[5];   // (32,)
    const float* norm_w  = (const float*)d_in[6];   // (128,)
    const float* W_out   = (const float*)d_in[7];   // (2048,4096)
    float* out = (float*)d_out;                     // (1,1024,2048) f32

    float* ws = (float*)d_ws;
    float* qkv_buf = ws;  ws += (size_t)S_LEN * CONVD;
    float* zba_buf = ws;  ws += (size_t)S_LEN * ZBAD;
    float* act_buf = ws;  ws += (size_t)S_LEN * CONVD;
    float* qn      = ws;  ws += (size_t)S_LEN * NKH * DKD;
    float* kn      = ws;  ws += (size_t)S_LEN * NKH * DKD;
    float* egb     = ws;  ws += (size_t)S_LEN * NVH;
    float* btb     = ws;  ws += (size_t)S_LEN * NVH;
    float* core    = ws;  ws += (size_t)S_LEN * VALD;
    float* gnb     = ws;  ws += (size_t)S_LEN * VALD;

    // 1) qkv = h @ W_qkv^T   (1024 x 8192, K=2048)
    gemm_bf16_wmma<<<dim3(CONVD / BN, S_LEN / BM), 256, 0, stream>>>(
        hidden, W_qkv, qkv_buf, S_LEN, CONVD, H_DIM);
    // 2) zba = h @ W_zba^T   (1024 x 4160, K=2048)
    gemm_bf16_wmma<<<dim3((ZBAD + BN - 1) / BN, S_LEN / BM), 256, 0, stream>>>(
        hidden, W_zba, zba_buf, S_LEN, ZBAD, H_DIM);
    // 3) causal conv K=4 + SiLU
    conv_silu_kernel<<<(S_LEN * CONVD / 4) / 256, 256, 0, stream>>>(
        qkv_buf, conv_w, act_buf);
    // 4) q/k L2 norm (+ DK^-0.5 into q)
    qk_norm_kernel<<<dim3(S_LEN, 2 * NKH), 128, 0, stream>>>(act_buf, qn, kn);
    // 5) gates
    gate_kernel<<<(S_LEN * NVH) / 256, 256, 0, stream>>>(
        zba_buf, A_log, dt_bias, egb, btb);
    // 6) sequential delta-rule scan (32 heads x 4 v-chunks = 128 blocks)
    scan_kernel<<<dim3(NVH, 4), 128, 0, stream>>>(qn, kn, act_buf, egb, btb, core);
    // 7) gated RMSNorm
    gated_norm_kernel<<<dim3(S_LEN, NVH), 128, 0, stream>>>(
        core, zba_buf, norm_w, gnb);
    // 8) out = gn @ W_out^T  (1024 x 2048, K=4096)
    gemm_bf16_wmma<<<dim3(H_DIM / BN, S_LEN / BM), 256, 0, stream>>>(
        gnb, W_out, out, S_LEN, H_DIM, VALD);
}
